// PairwiseModalityAttention_429496730177
// MI455X (gfx1250) — matrix operand order repaired and checked by reading, not compiled, not run on hardware
//
#include <hip/hip_runtime.h>
#include <hip/hip_bf16.h>
#include <math.h>

// ---------------------------------------------------------------------------
// PairwiseModalityAttention fused WMMA implementation for gfx1250 (MI455X)
//   video(512), audio(128), imu(64); HIDDEN=256, HEADS=4, HEAD_DIM=64, B=65536
//   All GEMMs on v_wmma_f32_16x16x32_f16 (f16 in, f32 accum); weights
//   pre-swizzled to B-fragment layout in d_ws; feature tiles staged in LDS.
//   Compile-time dims per modality -> exact LDS sizing, fully unrolled K-loops,
//   register-cached A-fragments for small operands, and up-front staging of
//   both partner tiles so pair-1 load latency overlaps pair-0 compute.
// ---------------------------------------------------------------------------

typedef __attribute__((ext_vector_type(16))) _Float16 v16h;
typedef __attribute__((ext_vector_type(8)))  _Float16 v8h;
typedef __attribute__((ext_vector_type(8)))  float    v8f;

#define HIDDEN  256
#define NTILES  16
#define HEADS   4
#define SCALE_F 0.125f   // 1/sqrt(64)

__device__ __forceinline__ v8f wmma_f16(v16h a, v16h b, v8f c) {
  return __builtin_amdgcn_wmma_f32_16x16x32_f16(false, a, false, b, (short)0, c,
                                                false, false);
}

// W (d_in x 256 f32 row-major) -> f16 B-fragment layout:
//   n = nt*16 + (lane&15), k = kf*32 + (lane>>4)*8 + (half&7) + (half>>3)*16
__global__ void pma_swz_weights(const float* __restrict__ W,
                                _Float16* __restrict__ out, int d_in) {
  int idx = blockIdx.x * 256 + threadIdx.x;
  if (idx >= d_in * HIDDEN) return;
  int half = idx & 15;
  int lane = (idx >> 4) & 31;
  int nt   = (idx >> 9) & 15;
  int kf   = idx >> 13;
  int n = nt * 16 + (lane & 15);
  int k = kf * 32 + (lane >> 4) * 8 + (half & 7) + (half >> 3) * 16;
  out[idx] = (_Float16)W[(size_t)k * HIDDEN + n];
}

// A-frag (16x32 f16) from LDS row-major [16][d]:
//   m = lane&15; halves 0..7 -> K = kf*32 + hi*8 + i; halves 8..15 -> +16
__device__ __forceinline__ v16h lds_afrag(const _Float16* base, int d, int kf,
                                          int lane) {
  const int m = lane & 15, hi = lane >> 4;
  const _Float16* p = base + m * d + kf * 32 + hi * 8;
  union { v16h v; v8h h[2]; } u;
  u.h[0] = *(const v8h*)(p);
  u.h[1] = *(const v8h*)(p + 16);
  return u.v;
}

// B-frag from pre-swizzled weights: 16 contiguous halves per lane.
__device__ __forceinline__ v16h gl_bfrag(const _Float16* __restrict__ Wswz,
                                         int nt, int kf, int lane) {
  const _Float16* p = Wswz + (((size_t)(kf * 16 + nt) * 32 + lane) << 4);
  union { v16h v; v8h h[2]; } u;
  u.h[0] = *(const v8h*)(p);
  u.h[1] = *(const v8h*)(p + 8);
  return u.v;
}

__device__ __forceinline__ void stage_tile(const float* __restrict__ X, int d,
                                           int row0, _Float16* dst, int lane) {
  const float4* src = (const float4*)(X + (size_t)row0 * d);
  int nvec = (16 * d) >> 2;
  for (int i = lane; i < nvec; i += 32) {
    float4 v = src[i];
    int o = i << 2;
    dst[o + 0] = (_Float16)v.x;
    dst[o + 1] = (_Float16)v.y;
    dst[o + 2] = (_Float16)v.z;
    dst[o + 3] = (_Float16)v.w;
  }
}

struct PairP {
  const float* xb;
  const _Float16 *wq, *wk, *wv, *wo;
  const float *bq, *bk, *bv, *bo;
  float* maps;
  int j;
};

struct ModP {
  const float* xa;
  const float* mm;
  float* out;
  PairP p[2];
};

// ---------------------------------------------------------------------------
template <int DA, int DB>
__device__ __forceinline__ void process_pair(const _Float16* sXa,
                                             const _Float16* sXb,
                                             _Float16* sAP, float* sAgg,
                                             const PairP& pw, const float* mm,
                                             int row0, int lane) {
  constexpr int KFA = DA / 32, KFB = DB / 32;
  constexpr bool CA = (KFA <= 8);   // cache Xa frags in regs if small
  constexpr bool CB = (KFB <= 8);   // cache Xb frags in regs if small
  const int m16 = lane & 15, hi = lane >> 4;

  v16h fa[CA ? KFA : 1];
  v16h fb[CB ? KFB : 1];
  if constexpr (CA) {
#pragma unroll
    for (int kf = 0; kf < KFA; ++kf) fa[kf] = lds_afrag(sXa, DA, kf, lane);
  }
  if constexpr (CB) {
#pragma unroll
    for (int kf = 0; kf < KFB; ++kf) fb[kf] = lds_afrag(sXb, DB, kf, lane);
  }

  float bm[8];
#pragma unroll
  for (int r = 0; r < 8; ++r)
    bm[r] = mm[(size_t)(row0 + r + 8 * hi) * 3 + pw.j];

  // ---- scores per head: sum_n Q[m,n]*K[m,n] over head columns -------------
  float w[HEADS][8];
#pragma unroll 1
  for (int h = 0; h < HEADS; ++h) {
    float part[8];
#pragma unroll
    for (int r = 0; r < 8; ++r) part[r] = 0.0f;

#pragma unroll 1
    for (int nt = 4 * h; nt < 4 * h + 4; ++nt) {
      const float bqv = pw.bq[nt * 16 + m16];
      const float bkv = pw.bk[nt * 16 + m16];
      v8f qt, kt;
#pragma unroll
      for (int r = 0; r < 8; ++r) { qt[r] = bqv; kt[r] = bkv; }
#pragma unroll
      for (int kf = 0; kf < KFA; ++kf)
        qt = wmma_f16(CA ? fa[kf] : lds_afrag(sXa, DA, kf, lane),
                      gl_bfrag(pw.wq, nt, kf, lane), qt);
#pragma unroll
      for (int kf = 0; kf < KFB; ++kf)
        kt = wmma_f16(CB ? fb[kf] : lds_afrag(sXb, DB, kf, lane),
                      gl_bfrag(pw.wk, nt, kf, lane), kt);
#pragma unroll
      for (int r = 0; r < 8; ++r) part[r] += qt[r] * kt[r];
    }
    // reduce across the 16 lanes of each half (C layout: lane = N column)
#pragma unroll
    for (int r = 0; r < 8; ++r) {
      float v = part[r];
      v += __shfl_xor(v, 1);
      v += __shfl_xor(v, 2);
      v += __shfl_xor(v, 4);
      v += __shfl_xor(v, 8);
      float s = v * SCALE_F * bm[r];
      w[h][r] = 1.0f / (1.0f + expf(-s));
    }
  }

  // ---- attention maps (B, 4) ----------------------------------------------
  if (m16 == 0) {
#pragma unroll
    for (int r = 0; r < 8; ++r)
#pragma unroll
      for (int h = 0; h < HEADS; ++h)
        pw.maps[(size_t)(row0 + r + 8 * hi) * HEADS + h] = w[h][r];
  }

  // ---- V projection + gate -> sAP (f16) -----------------------------------
#pragma unroll 1
  for (int nt = 0; nt < NTILES; ++nt) {
    const float bvv = pw.bv[nt * 16 + m16];
    v8f vt;
#pragma unroll
    for (int r = 0; r < 8; ++r) vt[r] = bvv;
#pragma unroll
    for (int kf = 0; kf < KFB; ++kf)
      vt = wmma_f16(CB ? fb[kf] : lds_afrag(sXb, DB, kf, lane),
                    gl_bfrag(pw.wv, nt, kf, lane), vt);
    const int h = nt >> 2;
#pragma unroll
    for (int r = 0; r < 8; ++r)
      sAP[(r + 8 * hi) * HIDDEN + nt * 16 + m16] = (_Float16)(w[h][r] * vt[r]);
  }

  // ---- O projection (K=256 -> 8 frags, cached), masked accumulate ---------
  v16h fo[8];
#pragma unroll
  for (int kf = 0; kf < 8; ++kf) fo[kf] = lds_afrag(sAP, HIDDEN, kf, lane);
#pragma unroll 1
  for (int nt = 0; nt < NTILES; ++nt) {
    const float bov = pw.bo[nt * 16 + m16];
    v8f ot;
#pragma unroll
    for (int r = 0; r < 8; ++r) ot[r] = bov;
#pragma unroll
    for (int kf = 0; kf < 8; ++kf)
      ot = wmma_f16(fo[kf], gl_bfrag(pw.wo, nt, kf, lane), ot);
#pragma unroll
    for (int r = 0; r < 8; ++r)
      sAgg[(r + 8 * hi) * HIDDEN + nt * 16 + m16] += ot[r] * bm[r];
  }
}

// ---------------------------------------------------------------------------
template <int DA, int DB0, int DB1>
__global__ __launch_bounds__(32)
void pma_fused(ModP A) {
  __shared__ _Float16 sXa[16 * DA];
  __shared__ _Float16 sXb0[16 * DB0];
  __shared__ _Float16 sXb1[16 * DB1];
  __shared__ _Float16 sAP[16 * HIDDEN];
  __shared__ float    sAgg[16 * HIDDEN];
  __shared__ float    sCnt[16];

  const int lane = threadIdx.x;
  const int row0 = blockIdx.x * 16;

  __builtin_prefetch(A.p[0].wq, 0, 1);
  __builtin_prefetch(A.p[0].wo, 0, 1);

  // stage all feature tiles up front: pair-1 staging latency overlaps pair-0
  stage_tile(A.xa, DA, row0, sXa, lane);
  stage_tile(A.p[0].xb, DB0, row0, sXb0, lane);
  stage_tile(A.p[1].xb, DB1, row0, sXb1, lane);
  for (int i = lane; i < 16 * HIDDEN; i += 32) sAgg[i] = 0.0f;

  process_pair<DA, DB0>(sXa, sXb0, sAP, sAgg, A.p[0], A.mm, row0, lane);
  process_pair<DA, DB1>(sXa, sXb1, sAP, sAgg, A.p[1], A.mm, row0, lane);

  // ---- normalize by max(count, 1) and store -------------------------------
  if (lane < 16) {
    float c = A.mm[(size_t)(row0 + lane) * 3 + A.p[0].j] +
              A.mm[(size_t)(row0 + lane) * 3 + A.p[1].j];
    sCnt[lane] = fmaxf(c, 1.0f);
  }
  for (int i = lane; i < 16 * HIDDEN; i += 32) {
    int m = i >> 8;
    A.out[(size_t)(row0 + m) * HIDDEN + (i & 255)] = sAgg[i] / sCnt[m];
  }
}

// ---------------------------------------------------------------------------
extern "C" void kernel_launch(void* const* d_in, const int* in_sizes, int n_in,
                              void* d_out, int out_size, void* d_ws,
                              size_t ws_size, hipStream_t stream) {
  const float* feats[3] = {(const float*)d_in[0], (const float*)d_in[1],
                           (const float*)d_in[2]};
  const float* mm = (const float*)d_in[3];
  const int B = in_sizes[0] / 512;

  // pair order: v->a, v->i, a->v, a->i, i->v, i->a
  static const int PA[6] = {0, 0, 1, 1, 2, 2};
  static const int PB[6] = {1, 2, 0, 2, 0, 1};
  static const int DIM[3] = {512, 128, 64};

  float* out = (float*)d_out;
  float* maps_base = out + 3ull * B * HIDDEN;

  // ---- pre-swizzle all weights into d_ws as f16 B-fragments ---------------
  _Float16* ws = (_Float16*)d_ws;
  size_t off_wq[6], off_wk[6], off_wv[6], off_wo[6];
  size_t cur = 0;
  for (int p = 0; p < 6; ++p) {
    const int da = DIM[PA[p]], db = DIM[PB[p]];
    off_wq[p] = cur; cur += (size_t)da * HIDDEN;
    off_wk[p] = cur; cur += (size_t)db * HIDDEN;
    off_wv[p] = cur; cur += (size_t)db * HIDDEN;
    off_wo[p] = cur; cur += (size_t)HIDDEN * HIDDEN;
  }
  for (int p = 0; p < 6; ++p) {
    const int da = DIM[PA[p]], db = DIM[PB[p]];
    const float* Wq = (const float*)d_in[4 + p * 8 + 0];
    const float* Wk = (const float*)d_in[4 + p * 8 + 2];
    const float* Wv = (const float*)d_in[4 + p * 8 + 4];
    const float* Wo = (const float*)d_in[4 + p * 8 + 6];
    pma_swz_weights<<<(da * HIDDEN + 255) / 256, 256, 0, stream>>>(
        Wq, ws + off_wq[p], da);
    pma_swz_weights<<<(db * HIDDEN + 255) / 256, 256, 0, stream>>>(
        Wk, ws + off_wk[p], db);
    pma_swz_weights<<<(db * HIDDEN + 255) / 256, 256, 0, stream>>>(
        Wv, ws + off_wv[p], db);
    pma_swz_weights<<<(HIDDEN * HIDDEN + 255) / 256, 256, 0, stream>>>(
        Wo, ws + off_wo[p], 256);
  }

  // ---- fused attention: one templated launch per output modality ----------
  for (int a = 0; a < 3; ++a) {
    ModP args;
    args.xa = feats[a];
    args.mm = mm;
    args.out = out + (size_t)a * B * HIDDEN;
    for (int s = 0; s < 2; ++s) {
      const int p = a * 2 + s;
      PairP& pw = args.p[s];
      pw.xb = feats[PB[p]];
      pw.wq = ws + off_wq[p];
      pw.wk = ws + off_wk[p];
      pw.wv = ws + off_wv[p];
      pw.wo = ws + off_wo[p];
      pw.bq = (const float*)d_in[4 + p * 8 + 1];
      pw.bk = (const float*)d_in[4 + p * 8 + 3];
      pw.bv = (const float*)d_in[4 + p * 8 + 5];
      pw.bo = (const float*)d_in[4 + p * 8 + 7];
      pw.maps = maps_base + (size_t)p * B * HEADS;
      pw.j = PB[p];
    }
    if (a == 0)      pma_fused<512, 128, 64 ><<<B / 16, 32, 0, stream>>>(args);
    else if (a == 1) pma_fused<128, 512, 64 ><<<B / 16, 32, 0, stream>>>(args);
    else             pma_fused<64,  512, 128><<<B / 16, 32, 0, stream>>>(args);
  }
}
